// GAT_19421842113018
// MI455X (gfx1250) — matrix operand order repaired and checked by reading, not compiled, not run on hardware
//
#include <hip/hip_runtime.h>
#include <hip/hip_bf16.h>

// ---------------------------------------------------------------------------
// GAT forward for MI455X (gfx1250): bf16 WMMA GEMM + segment softmax/sum.
// E=500000 edges, NE=100000 nodes, NR=1000, D=U=128 (from reference).
// ---------------------------------------------------------------------------

typedef __attribute__((ext_vector_type(16))) __bf16 v16bf;
typedef __attribute__((ext_vector_type(8)))  float  v8f;

#define DIM 128          // D == U == 128
#define KDIM 384         // 3*D
#define WPK_WORDS (KDIM * DIM / 2)   // 24576 u32 (pair-packed bf16 W)

__device__ __forceinline__ unsigned f2bf1(float f) {
    unsigned u = __float_as_uint(f);
    unsigned r = u + 0x7FFFu + ((u >> 16) & 1u);   // round-to-nearest-even
    return r >> 16;
}
__device__ __forceinline__ unsigned pack2(float a, float b) {
    return f2bf1(a) | (f2bf1(b) << 16);
}

__device__ __forceinline__ void atomicMaxF(float* addr, float v) {
    if (v >= 0.0f) atomicMax((int*)addr, __float_as_int(v));
    else           atomicMin((unsigned*)addr, __float_as_uint(v));
}

// --- 1a. pack W (384x128 f32) into bf16x2 words, WMMA-fragment-major --------
// Wshuf[((kt*8+nt)*32+lane)*8 + v] packs W[k,n],W[k+1,n] with
//   k = kt*32 + (lane>>4)*16 + 2v,  n = nt*16 + (lane&15)
// so each lane's B fragment is 32 contiguous bytes (2 x ds_load_b128).
__global__ __launch_bounds__(128) void k_prepW(const float* __restrict__ W,
                                               unsigned* __restrict__ Wshuf) {
    int j = blockIdx.x * 128 + threadIdx.x;          // 0 .. 24575
    if (j >= WPK_WORDS) return;
    int v    = j & 7;
    int lane = (j >> 3) & 31;
    int nt   = (j >> 8) & 7;
    int kt   = j >> 11;
    int kh = lane >> 4, lm = lane & 15;
    int p = kt * 16 + kh * 8 + v;                    // K pair index
    int n = nt * 16 + lm;
    Wshuf[j] = pack2(W[(2 * p) * DIM + n], W[(2 * p + 1) * DIM + n]);
}

// --- 1b. wa = W @ a  (384 floats) ------------------------------------------
__global__ __launch_bounds__(128) void k_wa(const float* __restrict__ W,
                                            const float* __restrict__ a,
                                            float* __restrict__ wa) {
    int k = blockIdx.x * 128 + threadIdx.x;
    if (k >= KDIM) return;
    const float4* row = (const float4*)(W + (size_t)k * DIM);
    const float4* av  = (const float4*)a;
    float s = 0.0f;
#pragma unroll 8
    for (int i = 0; i < DIM / 4; ++i) {
        float4 x = row[i], y = av[i];
        s += x.x * y.x + x.y * y.y + x.z * y.z + x.w * y.w;
    }
    wa[k] = s;
}

// --- 1c. fp32 table -> bf16 table (4 elements / thread) ---------------------
__global__ __launch_bounds__(256) void k_cvt(const float* __restrict__ src,
                                             unsigned* __restrict__ dst,
                                             int n4) {                 // n/4
    int i = blockIdx.x * 256 + threadIdx.x;
    if (i >= n4) return;
    float4 f = ((const float4*)src)[i];
    dst[i * 2]     = pack2(f.x, f.y);
    dst[i * 2 + 1] = pack2(f.z, f.w);
}

// --- 2. init output / segMax / denom ---------------------------------------
__global__ __launch_bounds__(256) void k_init(float* __restrict__ out,
                                              float* __restrict__ segMax,
                                              float* __restrict__ denom,
                                              int NE, int nOut) {
    int i = blockIdx.x * 256 + threadIdx.x;
    if (i < nOut) out[i] = 0.0f;
    if (i < NE) {
        segMax[i] = __uint_as_float(0xFF800000u);    // -inf
        denom[i]  = 0.0f;
    }
}

// --- 3. per-edge attention score + segment max ------------------------------
__global__ __launch_bounds__(128) void k_score(const int* __restrict__ hI,
                                               const int* __restrict__ rI,
                                               const int* __restrict__ tI,
                                               const float* __restrict__ ent,
                                               const float* __restrict__ rel,
                                               const float* __restrict__ wa,
                                               const float* __restrict__ ab,
                                               float* __restrict__ scoreOut,
                                               float* __restrict__ segMax,
                                               int E) {
    int e = blockIdx.x * 128 + threadIdx.x;
    if (e >= E) return;
    int h = hI[e], r = rI[e], t = tI[e];
    const float4* ph = (const float4*)(ent + (size_t)h * DIM);
    const float4* pr = (const float4*)(rel + (size_t)r * DIM);
    const float4* pt = (const float4*)(ent + (size_t)t * DIM);
    const float4* w  = (const float4*)wa;
    float s = ab[0];
#pragma unroll 8
    for (int i = 0; i < DIM / 4; ++i) {
        float4 x = ph[i], y = w[i];
        s += x.x * y.x + x.y * y.y + x.z * y.z + x.w * y.w;
    }
#pragma unroll 8
    for (int i = 0; i < DIM / 4; ++i) {
        float4 x = pr[i], y = w[32 + i];
        s += x.x * y.x + x.y * y.y + x.z * y.z + x.w * y.w;
    }
#pragma unroll 8
    for (int i = 0; i < DIM / 4; ++i) {
        float4 x = pt[i], y = w[64 + i];
        s += x.x * y.x + x.y * y.y + x.z * y.z + x.w * y.w;
    }
    s = (s >= 0.0f) ? s : 0.2f * s;                  // leaky(leaky(.))
    s = (s >= 0.0f) ? s : 0.2f * s;
    scoreOut[e] = s;
    atomicMaxF(&segMax[h], s);
}

// --- 4. exp(score - max) + segment denom ------------------------------------
__global__ __launch_bounds__(128) void k_exp(const int* __restrict__ hI,
                                             float* __restrict__ expS,
                                             const float* __restrict__ segMax,
                                             float* __restrict__ denom,
                                             int E) {
    int e = blockIdx.x * 128 + threadIdx.x;
    if (e >= E) return;
    int h = hI[e];
    float v = __expf(expS[e] - segMax[h]);
    expS[e] = v;
    atomicAdd(&denom[h], v);
}

// --- 5. WMMA GEMM (feats = X @ W, bf16 in / fp32 acc) + alpha + seg-sum -----
// 256 threads = 8 wave32; each wave owns one 16-edge M-tile per iteration and
// computes all 128 output columns (8 N-tiles), K=384 = 12 bf16 K-tiles.
// Fragment-major packed W (96 KB bf16) staged once per block into LDS;
// A fragments are direct b128 loads from the pre-converted bf16 tables.
__global__ __launch_bounds__(256) void k_gemm(const int* __restrict__ hI,
                                              const int* __restrict__ rI,
                                              const int* __restrict__ tI,
                                              const unsigned short* __restrict__ entB,
                                              const unsigned short* __restrict__ relB,
                                              const unsigned* __restrict__ Wshuf,
                                              const float* __restrict__ expS,
                                              const float* __restrict__ denom,
                                              float* __restrict__ out,
                                              int E, int ntiles) {
    extern __shared__ unsigned Wl[];                 // 24576 u32 = 96 KB

    const int tid = threadIdx.x;
    {   // cooperative LDS fill (uint4 copies)
        const uint4* src = (const uint4*)Wshuf;
        uint4* dst = (uint4*)Wl;
        for (int i = tid; i < WPK_WORDS / 4; i += 256) dst[i] = src[i];
    }
    __syncthreads();
    const uint4* Wl4 = (const uint4*)Wl;

    const int w    = tid >> 5;                       // wave id 0..7
    const int lane = tid & 31;
    const int lm   = lane & 15;                      // M row (A) / N col (B,D)
    const int kh   = lane >> 4;                      // K-half selector

    const v8f vzero = {0.f, 0.f, 0.f, 0.f, 0.f, 0.f, 0.f, 0.f};

    for (int tile = blockIdx.x * 8 + w; tile < ntiles; tile += gridDim.x * 8) {
        // Compiler barrier: prevents LICM from hoisting all 96 loop-invariant
        // B-fragment LDS loads out of the tile loop (which spills to scratch).
        asm volatile("" ::: "memory");

        const int eMine = tile * 16 + lm;
        const bool valid = (eMine < E);
        const int eC = valid ? eMine : (E - 1);

        const int h = hI[eC];
        const int r = rI[eC];
        const int t = tI[eC];
        const unsigned short* ph = entB + (size_t)h * DIM;
        const unsigned short* pr = relB + (size_t)r * DIM;
        const unsigned short* pt = entB + (size_t)t * DIM;

        const float myAlpha = valid ? (expS[eC] / denom[h]) : 0.0f;
        const int   myBase  = h * DIM;

        v8f c[8];
#pragma unroll
        for (int i = 0; i < 8; ++i) c[i] = vzero;

        for (int kt = 0; kt < 12; ++kt) {
            const unsigned short* p =
                (kt < 4 ? ph : (kt < 8 ? pr : pt)) + ((kt & 3) * 32);

            // A fragment (16x32 bf16, contiguous k order): per lane two b128 loads.
            // lanes 0-15 (kh=0): K 0..15 ; lanes 16-31: K 16..31
            union { uint4 q[2]; v16bf v; } A;
            A.q[0] = *(const uint4*)(p + kh * 16);
            A.q[1] = *(const uint4*)(p + kh * 16 + 8);

#pragma unroll
            for (int nt = 0; nt < 8; ++nt) {
                // B fragment: 32 contiguous bytes per lane in LDS.
                union { uint4 q[2]; v16bf v; } B;
                const int fb = ((kt * 8 + nt) * 32 + lane) * 2;
                B.q[0] = Wl4[fb];
                B.q[1] = Wl4[fb + 1];
                c[nt] = __builtin_amdgcn_wmma_f32_16x16x32_bf16(
                            false, A.v, false, B.v, (short)0, c[nt], false, false);
            }
        }

        // D layout: lane -> N = lm; VGPR v -> M = kh*8 + v. Broadcast alpha
        // and output row-base for the 8 D-rows this lane stores.
        float aV[8]; int bV[8];
#pragma unroll
        for (int v2 = 0; v2 < 8; ++v2) {
            int m = kh * 8 + v2;
            aV[v2] = __shfl(myAlpha, m, 32);
            bV[v2] = __shfl(myBase,  m, 32);
        }

#pragma unroll
        for (int nt = 0; nt < 8; ++nt) {
#pragma unroll
            for (int v2 = 0; v2 < 8; ++v2) {
                float val = c[nt][v2] * aV[v2];
                atomicAdd(out + bV[v2] + nt * 16 + lm, val);
            }
        }
    }
}

// --- 6. out = relu(out + bias) ----------------------------------------------
__global__ __launch_bounds__(256) void k_final(float* __restrict__ out,
                                               const float* __restrict__ bias,
                                               int nOut) {
    int i = blockIdx.x * 256 + threadIdx.x;
    if (i >= nOut) return;
    out[i] = fmaxf(out[i] + bias[i & (DIM - 1)], 0.0f);
}

// ---------------------------------------------------------------------------
extern "C" void kernel_launch(void* const* d_in, const int* in_sizes, int n_in,
                              void* d_out, int out_size, void* d_ws, size_t ws_size,
                              hipStream_t stream) {
    const int*   hI   = (const int*)d_in[0];
    const int*   rI   = (const int*)d_in[1];
    const int*   tI   = (const int*)d_in[2];
    const float* ent  = (const float*)d_in[3];
    const float* rel  = (const float*)d_in[4];
    const float* W    = (const float*)d_in[5];
    const float* a    = (const float*)d_in[6];
    const float* ab   = (const float*)d_in[7];
    const float* bias = (const float*)d_in[8];

    const int E   = in_sizes[0];
    const int NE  = in_sizes[3] / DIM;
    const int NR  = in_sizes[4] / DIM;
    float* out = (float*)d_out;

    // workspace layout (16B-aligned sections)
    char* ws = (char*)d_ws;
    unsigned*       Wshuf  = (unsigned*)ws;                              // 98304 B
    float*          wa     = (float*)(ws + 98304);                       // 1536 B
    float*          segMax = (float*)(ws + 99840);                       // NE*4
    float*          denom  = (float*)(ws + 99840 + (size_t)NE * 4);      // NE*4
    float*          expS   = (float*)(ws + 99840 + (size_t)NE * 8);      // E*4
    size_t off = 99840 + (size_t)NE * 8 + (size_t)E * 4;
    off = (off + 15) & ~(size_t)15;
    unsigned short* entB   = (unsigned short*)(ws + off);                // NE*128*2
    off += (size_t)NE * DIM * 2;
    unsigned short* relB   = (unsigned short*)(ws + off);                // NR*128*2

    const int nOut = NE * DIM;

    hipLaunchKernelGGL(k_prepW, dim3((WPK_WORDS + 127) / 128), dim3(128), 0, stream,
                       W, Wshuf);
    hipLaunchKernelGGL(k_wa, dim3((KDIM + 127) / 128), dim3(128), 0, stream,
                       W, a, wa);
    {   // fp32 -> bf16 table conversions
        int n4e = NE * DIM / 4, n4r = NR * DIM / 4;
        hipLaunchKernelGGL(k_cvt, dim3((n4e + 255) / 256), dim3(256), 0, stream,
                           ent, (unsigned*)entB, n4e);
        hipLaunchKernelGGL(k_cvt, dim3((n4r + 255) / 256), dim3(256), 0, stream,
                           rel, (unsigned*)relB, n4r);
    }
    hipLaunchKernelGGL(k_init, dim3((nOut + 255) / 256), dim3(256), 0, stream,
                       out, segMax, denom, NE, nOut);
    hipLaunchKernelGGL(k_score, dim3((E + 127) / 128), dim3(128), 0, stream,
                       hI, rI, tI, ent, rel, wa, ab, expS, segMax, E);
    hipLaunchKernelGGL(k_exp, dim3((E + 127) / 128), dim3(128), 0, stream,
                       hI, expS, segMax, denom, E);

    const int ntiles = (E + 15) / 16;
    int blocks = (ntiles + 7) / 8;
    if (blocks > 1024) blocks = 1024;
    hipLaunchKernelGGL(k_gemm, dim3(blocks), dim3(256), 98304, stream,
                       hI, rI, tI, entB, relB, Wshuf, expS, denom, out, E, ntiles);

    hipLaunchKernelGGL(k_final, dim3((nOut + 255) / 256), dim3(256), 0, stream,
                       out, bias, nOut);
}
